// Encoder_38963943309348
// MI455X (gfx1250) — matrix operand order repaired and checked by reading, not compiled, not run on hardware
//
#include <hip/hip_runtime.h>
#include <hip/hip_bf16.h>

typedef __attribute__((ext_vector_type(16))) _Float16 v16h;
typedef __attribute__((ext_vector_type(8)))  _Float16 v8h;
typedef __attribute__((ext_vector_type(8)))  float    v8f;

#define D 128
#define LDSTR 136   // padded LDS row stride in halves; 272B per row (16B aligned)

static __device__ __forceinline__ int lane_id() { return (int)(threadIdx.x & 31); }

// ---- WMMA fragment loaders: vectorized (2x ds_load_b128 each) --------------

// A fragment 16x32 f16 from row-major LDS tile; base -> (row0, k0); stride in halves.
// lane l: m=l&15, h=l>>4; needs K = [8h, 8h+8) and [16+8h, 16+8h+8) of row m,
// packed in exactly that order -> two contiguous 16B runs.
static __device__ __forceinline__ v16h frag_a_vec(const _Float16* base, int stride) {
  int l = lane_id(); int m = l & 15; int h = l >> 4;
  const _Float16* p = base + m * stride + 8 * h;
  v8h lo = *(const v8h*)(p);
  v8h hi = *(const v8h*)(p + 16);
  union { v16h v; v8h h2[2]; } u;
  u.h2[0] = lo; u.h2[1] = hi;
  return u.v;
}

// B fragment 32x16 f16 from a TRANSPOSED (N-major) LDS tile; baseT -> (n0, k0);
// lane l: n=l&15, h=l>>4; needs K = [8h, 8h+8) and [16+8h, 16+8h+8) of column n
// -> two contiguous 16B runs in the transposed tile.
static __device__ __forceinline__ v16h frag_b_vecT(const _Float16* baseT, int stride) {
  int l = lane_id(); int n = l & 15; int h = l >> 4;
  const _Float16* p = baseT + n * stride + 8 * h;
  v8h lo = *(const v8h*)(p);
  v8h hi = *(const v8h*)(p + 16);
  union { v16h v; v8h h2[2]; } u;
  u.h2[0] = lo; u.h2[1] = hi;
  return u.v;
}

// ---- utility kernels -------------------------------------------------------

__global__ void k_zero_f32(float* p, long long n) {
  long long i = (long long)blockIdx.x * blockDim.x + threadIdx.x;
  long long s = (long long)gridDim.x * blockDim.x;
  for (; i < n; i += s) p[i] = 0.0f;
}

__global__ void k_degree(const int* ei, int E2, int EH, float* deg_in, float* deg_out) {
  int i = blockIdx.x * blockDim.x + threadIdx.x, s = gridDim.x * blockDim.x;
  for (int e = i; e < E2; e += s) {
    float* deg = (e < EH) ? deg_in : deg_out;
    atomicAdd(&deg[ei[e]], 1.0f);
  }
}

__global__ void k_deginv(float* deg, long long n) {
  long long i = (long long)blockIdx.x * blockDim.x + threadIdx.x;
  long long s = (long long)gridDim.x * blockDim.x;
  for (; i < n; i += s) {
    float d = deg[i];
    deg[i] = (d > 0.0f) ? rsqrtf(fmaxf(d, 1.0f)) : 0.0f;
  }
}

__global__ void k_hist(const int* et, int E2, int EH, int NREL, int* hist) {
  int i = blockIdx.x * blockDim.x + threadIdx.x, s = gridDim.x * blockDim.x;
  for (int e = i; e < E2; e += s)
    atomicAdd(&hist[((e >= EH) ? NREL : 0) + et[e]], 1);
}

__global__ void k_scan(const int* hist, int* offsets, int* cursor, int G) {
  if (threadIdx.x == 0 && blockIdx.x == 0) {
    int run = 0;
    for (int k = 0; k < G; ++k) { offsets[k] = run; cursor[k] = run; run += hist[k]; }
    offsets[G] = run;
  }
}

__global__ void k_scatter(const int* et, int E2, int EH, int NREL, int* cursor, int* sorted) {
  int i = blockIdx.x * blockDim.x + threadIdx.x, s = gridDim.x * blockDim.x;
  for (int e = i; e < E2; e += s) {
    int key = ((e >= EH) ? NREL : 0) + et[e];
    int pos = atomicAdd(&cursor[key], 1);
    sorted[pos] = e;
  }
}

__global__ void k_x2h(const float* x, _Float16* xh, long long n) {
  long long i = (long long)blockIdx.x * blockDim.x + threadIdx.x;
  long long s = (long long)gridDim.x * blockDim.x;
  for (; i < n; i += s) xh[i] = (_Float16)x[i];
}

// ---- Mt[g] = transpose( C(rel_g) @ W_g )  (f16, N-major) -------------------
__global__ void k_build_m(const float* rel_embed, const float* loop_rel,
                          const float* w_in, const float* w_out, const float* w_loop,
                          _Float16* Mt, int NREL) {
  __shared__ __align__(16) _Float16 relh[D];
  __shared__ __align__(16) _Float16 Cl[D * LDSTR];   // circulant, row-major (m,k)
  __shared__ __align__(16) _Float16 Wt[D * LDSTR];   // W transposed (n,k)
  int g = blockIdx.x;
  int r; const float* W;
  if (g < NREL)          { r = g;        W = w_in;  }
  else if (g < 2 * NREL) { r = g - NREL; W = w_out; }
  else                   { r = NREL;     W = w_loop; }
  const float* relrow = (r < NREL) ? (rel_embed + (long long)r * D) : loop_rel;
  int tid = threadIdx.x;
  if (tid < D) relh[tid] = (_Float16)relrow[tid];
  __syncthreads();
  for (int i = tid; i < D * D; i += blockDim.x) {
    int m = i >> 7, k = i & 127;
    Cl[m * LDSTR + k] = relh[(m + k) & (D - 1)];   // C[m][k] = rel[(m+k) mod 128]
    Wt[k * LDSTR + m] = (_Float16)W[i];            // Wt[n=k? no] -- see note below
  }
  __syncthreads();
  // NOTE on Wt: W is row-major (k, n); W[i] with m=i>>7 (k-index) and k=i&127 (n-index)
  // gives Wt[n * LDSTR + k] = W[k * D + n]  ==  Wt[(i&127) * LDSTR + (i>>7)] = W[i].
  // The line above stores exactly that (variable names reused).

  int n0 = (tid >> 5) * 16;                 // one 16-col tile per wave
  _Float16* out = Mt + (long long)g * D * D;  // transposed: out[n*D + m]
  for (int m0 = 0; m0 < D; m0 += 16) {
    v8f acc = {};
#pragma unroll
    for (int k0 = 0; k0 < D; k0 += 32) {
      v16h a = frag_a_vec(Cl + m0 * LDSTR + k0, LDSTR);
      v16h b = frag_b_vecT(Wt + n0 * LDSTR + k0, LDSTR);
      acc = __builtin_amdgcn_wmma_f32_16x16x32_f16(false, a, false, b, (short)0, acc, false, false);
    }
    int l = lane_id(); int n = l & 15; int h = l >> 4;
    v8h dv;
#pragma unroll
    for (int j = 0; j < 8; ++j) dv[j] = (_Float16)acc[j];   // m = m0 + 8h + j
    *(v8h*)&out[(long long)(n0 + n) * D + m0 + 8 * h] = dv; // contiguous 16B store
  }
}

// ---- main propagate: per relation-group GEMM, scatter-add ------------------
__global__ void k_prop(const _Float16* xh, const _Float16* Mt,
                       const int* ei, const int* sorted, const int* offsets,
                       const float* deg_in, const float* deg_out,
                       float* accum, int NREL, int E2) {
  __shared__ __align__(16) _Float16 Mg[D * LDSTR];   // transposed tile (n,k)
  __shared__ __align__(16) _Float16 Xl[16 * LDSTR];  // row tile (m,k)
  __shared__ int srcl[16], dstl[16];
  __shared__ float norml[16];

  int g = blockIdx.x;
  int beg = offsets[g], nedges = offsets[g + 1] - beg;
  if ((int)blockIdx.y * 16 >= nedges) return;

  int tid = threadIdx.x;
  const _Float16* Msrc = Mt + (long long)g * D * D;
  for (int i = tid * 8; i < D * D; i += blockDim.x * 8) {
    int n = i >> 7, k = i & 127;
    *(v8h*)&Mg[n * LDSTR + k] = *(const v8h*)&Msrc[i];
  }
  // prefetch next relation's tile into cache while we compute (g+1 <= 2*NREL valid)
  __builtin_prefetch(Msrc + (long long)D * D + (size_t)tid * 64, 0, 1);
  const float* dinv = (g < NREL) ? deg_in : deg_out;

  int ntiles = (nedges + 15) >> 4;
  for (int t = blockIdx.y; t < ntiles; t += gridDim.y) {
    __syncthreads();
    if (tid < 16) {
      int idx = t * 16 + tid;
      if (idx < nedges) {
        int e = sorted[beg + idx];
        int s = ei[e], d = ei[E2 + e];
        srcl[tid] = s; dstl[tid] = d;
        norml[tid] = dinv[s] * dinv[d];
      } else { srcl[tid] = 0; dstl[tid] = 0; norml[tid] = 0.0f; }
    }
    __syncthreads();
    for (int i = tid * 8; i < 16 * D; i += blockDim.x * 8) {
      int m = i >> 7, c = i & 127;
      v8h xv = *(const v8h*)&xh[(long long)srcl[m] * D + c];
      float nm = norml[m];
      v8h sv;
#pragma unroll
      for (int j = 0; j < 8; ++j) sv[j] = (_Float16)((float)xv[j] * nm);
      *(v8h*)&Xl[m * LDSTR + c] = sv;
    }
    __syncthreads();

    int n0 = (tid >> 5) * 16;
    v8f acc = {};
#pragma unroll
    for (int k0 = 0; k0 < D; k0 += 32) {
      v16h a = frag_a_vec(Xl + k0, LDSTR);
      v16h b = frag_b_vecT(Mg + n0 * LDSTR + k0, LDSTR);
      acc = __builtin_amdgcn_wmma_f32_16x16x32_f16(false, a, false, b, (short)0, acc, false, false);
    }
    int l = lane_id(); int n = l & 15; int h = l >> 4;
#pragma unroll
    for (int j = 0; j < 8; ++j)
      atomicAdd(&accum[(long long)dstl[j + 8 * h] * D + n0 + n], acc[j]);
  }
}

// ---- self-loop: accum += x @ M_loop ----------------------------------------
__global__ void k_loop(const _Float16* xh, const _Float16* Mt, float* accum,
                       int NREL, int NENT) {
  __shared__ __align__(16) _Float16 Mg[D * LDSTR];
  __shared__ __align__(16) _Float16 Xl[16 * LDSTR];
  int tid = threadIdx.x;
  const _Float16* Msrc = Mt + (long long)(2 * NREL) * D * D;
  for (int i = tid * 8; i < D * D; i += blockDim.x * 8) {
    int n = i >> 7, k = i & 127;
    *(v8h*)&Mg[n * LDSTR + k] = *(const v8h*)&Msrc[i];
  }
  int ntiles = (NENT + 15) >> 4;
  for (int t = blockIdx.x; t < ntiles; t += gridDim.x) {
    __syncthreads();
    int row0 = t * 16;
    for (int i = tid * 8; i < 16 * D; i += blockDim.x * 8) {
      int m = i >> 7, c = i & 127, r = row0 + m;
      v8h xv = {};
      if (r < NENT) xv = *(const v8h*)&xh[(long long)r * D + c];
      *(v8h*)&Xl[m * LDSTR + c] = xv;
    }
    __syncthreads();
    int n0 = (tid >> 5) * 16;
    v8f acc = {};
#pragma unroll
    for (int k0 = 0; k0 < D; k0 += 32) {
      v16h a = frag_a_vec(Xl + k0, LDSTR);
      v16h b = frag_b_vecT(Mg + n0 * LDSTR + k0, LDSTR);
      acc = __builtin_amdgcn_wmma_f32_16x16x32_f16(false, a, false, b, (short)0, acc, false, false);
    }
    int l = lane_id(); int n = l & 15; int h = l >> 4;
#pragma unroll
    for (int j = 0; j < 8; ++j) {
      int r = row0 + j + 8 * h;
      if (r < NENT) accum[(long long)r * D + n0 + n] += acc[j];
    }
  }
}

// ---- batch norm: scale+bias, column sums, normalize ------------------------
__global__ void k_stats(float* accum, const float* bias, float* colsum, float* colsum2,
                        long long total) {
  __shared__ float s1[D], s2[D];
  int tid = threadIdx.x;
  if (tid < D) { s1[tid] = 0.0f; s2[tid] = 0.0f; }
  __syncthreads();
  long long i = (long long)blockIdx.x * blockDim.x + tid;
  long long s = (long long)gridDim.x * blockDim.x;
  for (; i < total; i += s) {
    int c = (int)(i & (D - 1));
    float v = accum[i] * (1.0f / 3.0f) + bias[c];
    accum[i] = v;
    atomicAdd(&s1[c], v);
    atomicAdd(&s2[c], v * v);
  }
  __syncthreads();
  if (tid < D) { atomicAdd(&colsum[tid], s1[tid]); atomicAdd(&colsum2[tid], s2[tid]); }
}

__global__ void k_mv(float* colsum, float* colsum2, int NENT) {
  int c = threadIdx.x;
  if (c < D) {
    float mu  = colsum[c] / (float)NENT;
    float var = colsum2[c] / (float)NENT - mu * mu;
    colsum[c]  = mu;
    colsum2[c] = rsqrtf(var + 1e-5f);
  }
}

__global__ void k_norm(float* accum, const float* mu, const float* ivar,
                       const float* bnw, const float* bnb, long long total) {
  long long i = (long long)blockIdx.x * blockDim.x + threadIdx.x;
  long long s = (long long)gridDim.x * blockDim.x;
  for (; i < total; i += s) {
    int c = (int)(i & (D - 1));
    accum[i] = (accum[i] - mu[c]) * ivar[c] * bnw[c] + bnb[c];
  }
}

// ---- rel_out = rel_embed @ w_rel ------------------------------------------
__global__ void k_relout(const float* rel_embed, const float* w_rel, float* out, int NREL) {
  __shared__ __align__(16) _Float16 Wt[D * LDSTR];   // w_rel transposed (n,k)
  __shared__ __align__(16) _Float16 Xl[16 * LDSTR];
  int tid = threadIdx.x;
  for (int i = tid; i < D * D; i += blockDim.x) {
    int k = i >> 7, n = i & 127;
    Wt[n * LDSTR + k] = (_Float16)w_rel[i];
  }
  int ntiles = (NREL + 15) >> 4;
  for (int t = blockIdx.x; t < ntiles; t += gridDim.x) {
    __syncthreads();
    int row0 = t * 16;
    for (int i = tid; i < 16 * D; i += blockDim.x) {
      int m = i >> 7, c = i & 127, r = row0 + m;
      Xl[m * LDSTR + c] = (r < NREL) ? (_Float16)rel_embed[(long long)r * D + c]
                                     : (_Float16)0.0f;
    }
    __syncthreads();
    int n0 = (tid >> 5) * 16;
    v8f acc = {};
#pragma unroll
    for (int k0 = 0; k0 < D; k0 += 32) {
      v16h a = frag_a_vec(Xl + k0, LDSTR);
      v16h b = frag_b_vecT(Wt + n0 * LDSTR + k0, LDSTR);
      acc = __builtin_amdgcn_wmma_f32_16x16x32_f16(false, a, false, b, (short)0, acc, false, false);
    }
    int l = lane_id(); int n = l & 15; int h = l >> 4;
#pragma unroll
    for (int j = 0; j < 8; ++j) {
      int r = row0 + j + 8 * h;
      if (r < NREL) out[(long long)r * D + n0 + n] = acc[j];
    }
  }
}

// ---- host launcher ---------------------------------------------------------
extern "C" void kernel_launch(void* const* d_in, const int* in_sizes, int n_in,
                              void* d_out, int out_size, void* d_ws, size_t ws_size,
                              hipStream_t stream) {
  (void)n_in; (void)out_size; (void)ws_size;
  const float* x         = (const float*)d_in[0];
  const float* rel_embed = (const float*)d_in[1];
  const float* w_in      = (const float*)d_in[2];
  const float* w_out     = (const float*)d_in[3];
  const float* w_loop    = (const float*)d_in[4];
  const float* w_rel     = (const float*)d_in[5];
  const float* loop_rel  = (const float*)d_in[6];
  const float* bias      = (const float*)d_in[7];
  const float* bn_w      = (const float*)d_in[8];
  const float* bn_b      = (const float*)d_in[9];
  const int*   ei        = (const int*)d_in[10];
  const int*   et        = (const int*)d_in[11];

  const int NENT = in_sizes[0] / D;
  const int NREL = in_sizes[1] / D;
  const int E2   = in_sizes[11];
  const int EH   = E2 / 2;
  const int G    = 2 * NREL;

  char* ws = (char*)d_ws;
  size_t off = 0;
  auto take = [&](size_t bytes) -> char* {
    char* p = ws + off;
    off += (bytes + 255) & ~(size_t)255;
    return p;
  };
  float*     deg     = (float*)take((size_t)2 * NENT * sizeof(float)); // in, then out
  int*       hist    = (int*)take((size_t)G * sizeof(int));
  int*       offsets = (int*)take((size_t)(G + 1) * sizeof(int));
  int*       cursor  = (int*)take((size_t)G * sizeof(int));
  int*       sorted  = (int*)take((size_t)E2 * sizeof(int));
  _Float16*  xh      = (_Float16*)take((size_t)NENT * D * sizeof(_Float16));
  _Float16*  Mt      = (_Float16*)take((size_t)(G + 1) * D * D * sizeof(_Float16));
  float*     colsum  = (float*)take(D * sizeof(float));
  float*     colsum2 = (float*)take(D * sizeof(float));

  float* accum  = (float*)d_out;              // entity output region doubles as accumulator
  float* relout = accum + (size_t)NENT * D;   // rel_out region

  k_zero_f32<<<256, 256, 0, stream>>>(deg, (long long)2 * NENT);
  k_zero_f32<<<8, 256, 0, stream>>>((float*)hist, (long long)G);
  k_zero_f32<<<1, 256, 0, stream>>>(colsum, (long long)D);
  k_zero_f32<<<1, 256, 0, stream>>>(colsum2, (long long)D);
  k_zero_f32<<<2048, 256, 0, stream>>>(accum, (long long)NENT * D);

  k_degree<<<2048, 256, 0, stream>>>(ei, E2, EH, deg, deg + NENT);
  k_deginv<<<512, 256, 0, stream>>>(deg, (long long)2 * NENT);
  k_hist<<<2048, 256, 0, stream>>>(et, E2, EH, NREL, hist);
  k_scan<<<1, 1, 0, stream>>>(hist, offsets, cursor, G);
  k_scatter<<<2048, 256, 0, stream>>>(et, E2, EH, NREL, cursor, sorted);
  k_x2h<<<2048, 256, 0, stream>>>(x, xh, (long long)NENT * D);
  k_build_m<<<G + 1, 256, 0, stream>>>(rel_embed, loop_rel, w_in, w_out, w_loop, Mt, NREL);
  k_prop<<<dim3(G, 32), 256, 0, stream>>>(xh, Mt, ei, sorted, offsets,
                                          deg, deg + NENT, accum, NREL, E2);
  k_loop<<<1024, 256, 0, stream>>>(xh, Mt, accum, NREL, NENT);
  k_stats<<<512, 256, 0, stream>>>(accum, bias, colsum, colsum2, (long long)NENT * D);
  k_mv<<<1, D, 0, stream>>>(colsum, colsum2, NENT);
  k_norm<<<2048, 256, 0, stream>>>(accum, colsum, colsum2, bn_w, bn_b, (long long)NENT * D);
  k_relout<<<32, 256, 0, stream>>>(rel_embed, w_rel, relout, NREL);
}
